// GNN_16836271800585
// MI455X (gfx1250) — matrix operand order repaired and checked by reading, not compiled, not run on hardware
//
#include <hip/hip_runtime.h>

// ---------------------------------------------------------------------------
// GraphSAGE 3-layer GNN for MI455X (gfx1250, wave32, WMMA)
// Edge phase: L2-resident gather/scatter with hw f32 atomics.
// Node phase: 16x16x32 f16 WMMA GEMM, f32 accumulate, fused bias+sigmoid.
// Rev 3: template NKT -> fully unrolled K loop (scheduler sees all WMMAs and
//        loads at once); fast-rcp sigmoid instead of IEEE divide.
// ---------------------------------------------------------------------------

#define GN_N 50000
#define GN_E 1000000
#define GN_D 64

typedef _Float16 half16 __attribute__((ext_vector_type(16)));
typedef _Float16 half8v __attribute__((ext_vector_type(8)));
typedef float    float8 __attribute__((ext_vector_type(8)));

// ---------------- elementwise helpers ----------------

__global__ void gnn_zero_f32(float* p, int n) {
    int t = blockIdx.x * blockDim.x + threadIdx.x;
    if (t < n) p[t] = 0.0f;
}

__global__ void gnn_edge_deg(const int* __restrict__ dst, float* __restrict__ deg, int E) {
    int e = blockIdx.x * blockDim.x + threadIdx.x;
    if (e >= E) return;
    unsafeAtomicAdd(deg + dst[e], 1.0f);
}

__global__ void gnn_invert_deg(float* deg, int n) {
    int t = blockIdx.x * blockDim.x + threadIdx.x;
    if (t < n) deg[t] = 1.0f / fmaxf(deg[t], 1.0f);
}

// Build padded layer-0 input: [features(3) | ones(7) | zeros(6)] -> 16 dims.
// Writes f32 (gather source) and f16 self-half of layer-0 A operand.
__global__ void gnn_build_h0(const float* __restrict__ f, float* __restrict__ h0,
                             _Float16* __restrict__ a0, int n) {
    int t = blockIdx.x * blockDim.x + threadIdx.x;
    if (t >= n * 16) return;
    int node = t >> 4, d = t & 15;
    float v = (d < 3) ? f[node * 3 + d] : ((d < 10) ? 1.0f : 0.0f);
    h0[t] = v;
    a0[(size_t)node * 32 + d] = (_Float16)v;
}

// Pack [Ws; Wn] (zero-padded) into WMMA B-fragment order:
// fragment (kt,nt): lane l holds col n = nt*16+(l&15), K = kt*32+(l>>4)*16+i.
__global__ void gnn_pack_w(const float* __restrict__ Ws, const float* __restrict__ Wn,
                           _Float16* __restrict__ wp, int halfK, int dreal, int K) {
    int t = blockIdx.x * blockDim.x + threadIdx.x;
    if (t >= K * 64) return;
    int k = t >> 6, n = t & 63;
    int kk; const float* W;
    if (k < halfK) { kk = k; W = Ws; } else { kk = k - halfK; W = Wn; }
    float v = (kk < dreal) ? W[kk * 64 + n] : 0.0f;
    int kt = k >> 5, rem = k & 31;
    int lane = ((rem >> 4) << 4) | (n & 15);
    int i = rem & 15;
    int nt = n >> 4;
    wp[(((size_t)(kt * 4 + nt) * 32 + lane) << 4) + i] = (_Float16)v;
}

// h_neigh = agg * deg_inv, packed f16 into neigh-half of the A operand.
__global__ void gnn_pack_neigh(const float* __restrict__ agg, const float* __restrict__ deginv,
                               _Float16* __restrict__ aout, int dim, int astride, int n) {
    int t = blockIdx.x * blockDim.x + threadIdx.x;
    if (t >= n * dim) return;
    int node = t / dim, d = t - node * dim;
    aout[(size_t)node * astride + dim + d] = (_Float16)(agg[t] * deginv[node]);
}

// ---------------- edge scatter: m = h[src]*ew, agg[dst] += m ----------------
// DIM/4 lanes per edge, 4 dims (float4) per lane -> coalesced row segments.
template <int DIM>
__global__ void gnn_edge_scatter(const int* __restrict__ src, const int* __restrict__ dst,
                                 const float* __restrict__ ew, const float* __restrict__ h,
                                 float* __restrict__ agg, int E) {
    const int CPE = DIM / 4;
    int t = blockIdx.x * blockDim.x + threadIdx.x;
    int e = t / CPE;
    if (e >= E) return;
    int c = (t - e * CPE) * 4;
    int s = src[e], d = dst[e];
    float w = ew[e];
    float4 v = *(const float4*)(h + (size_t)s * DIM + c);
    float* ap = agg + (size_t)d * DIM + c;
    unsafeAtomicAdd(ap + 0, v.x * w);
    unsafeAtomicAdd(ap + 1, v.y * w);
    unsafeAtomicAdd(ap + 2, v.z * w);
    unsafeAtomicAdd(ap + 3, v.w * w);
}

// ---------------- node GEMM: D[16x64] = A[16xK] * B[Kx64] + bias ----------------
// One wave per 16-node tile. NKT K-steps of v_wmma_f32_16x16x32_f16,
// fully unrolled at compile time so the scheduler can pipeline loads
// from step kt+1 behind the WMMAs of step kt.
template <int NKT, bool SIG>
__global__ void __launch_bounds__(256)
gnn_sage_gemm(const _Float16* __restrict__ A, int astride,
              const _Float16* __restrict__ wp, const float* __restrict__ bias,
              float* __restrict__ hout, _Float16* __restrict__ aout, int aostride,
              int ntiles) {
    int wave = (blockIdx.x * blockDim.x + threadIdx.x) >> 5;
    int lane = threadIdx.x & 31;
    if (wave >= ntiles) return;          // wave-uniform: EXEC stays all-ones below

    int ln15 = lane & 15;
    int khalf = lane >> 4;
    const _Float16* arow = A + (size_t)(wave * 16 + ln15) * astride;

    float8 c[4];
#pragma unroll
    for (int nt = 0; nt < 4; ++nt) {
        float bv = bias[nt * 16 + ln15];
        c[nt] = (float8){bv, bv, bv, bv, bv, bv, bv, bv};
    }

#pragma unroll
    for (int kt = 0; kt < NKT; ++kt) {
        int k0 = kt * 32 + khalf * 16;
        half8v lo = *(const half8v*)(arow + k0);        // A: K = k0..k0+7
        half8v hi = *(const half8v*)(arow + k0 + 8);    // A: K = k0+8..k0+15
        half16 b[4];
#pragma unroll
        for (int nt = 0; nt < 4; ++nt)
            b[nt] = *(const half16*)(wp + (((size_t)(kt * 4 + nt) * 32 + lane) << 4));
        half16 a;
#pragma unroll
        for (int i = 0; i < 8; ++i) { a[i] = lo[i]; a[8 + i] = hi[i]; }
#pragma unroll
        for (int nt = 0; nt < 4; ++nt) {
            c[nt] = __builtin_amdgcn_wmma_f32_16x16x32_f16(
                /*neg_a=*/false, a, /*neg_b=*/false, b[nt],
                /*c_mod=*/(short)0, c[nt], /*reuse_a=*/false, /*reuse_b=*/false);
        }
    }

    // C/D layout: VGPR r -> row r + 8*khalf, col = nt*16 + (lane&15)
#pragma unroll
    for (int nt = 0; nt < 4; ++nt) {
#pragma unroll
        for (int r = 0; r < 8; ++r) {
            float v = c[nt][r];
            if (SIG) {
                // fast sigmoid: v_mul + v_exp + v_add + v_rcp (vs IEEE divide)
                v = __builtin_amdgcn_rcpf(1.0f + __expf(-v));
            }
            int node = wave * 16 + r + 8 * khalf;
            int col = nt * 16 + ln15;
            hout[(size_t)node * 64 + col] = v;
            if (aout) aout[(size_t)node * aostride + col] = (_Float16)v;
        }
    }
}

// ---------------- readout: out = [features | h3] @ Wr + br ----------------
__global__ void gnn_readout(const float* __restrict__ f, const float* __restrict__ h,
                            const float* __restrict__ Wr, const float* __restrict__ br,
                            float* __restrict__ out, int n) {
    int node = blockIdx.x * blockDim.x + threadIdx.x;
    if (node >= n) return;
    float acc = br[0];
    acc += f[node * 3 + 0] * Wr[0] + f[node * 3 + 1] * Wr[1] + f[node * 3 + 2] * Wr[2];
    const float4* hp = (const float4*)(h + (size_t)node * 64);
#pragma unroll
    for (int i = 0; i < 16; ++i) {
        float4 v = hp[i];
        acc += v.x * Wr[3 + 4 * i] + v.y * Wr[4 + 4 * i] + v.z * Wr[5 + 4 * i] + v.w * Wr[6 + 4 * i];
    }
    out[node] = acc;
}

// ---------------------------------------------------------------------------

extern "C" void kernel_launch(void* const* d_in, const int* in_sizes, int n_in,
                              void* d_out, int out_size, void* d_ws, size_t ws_size,
                              hipStream_t stream) {
    const int N = GN_N, E = GN_E;
    const float* features = (const float*)d_in[0];
    const float* e_feat   = (const float*)d_in[1];
    const int*   src      = (const int*)d_in[2];
    const int*   dst      = (const int*)d_in[3];
    const float* Ws0 = (const float*)d_in[4];
    const float* Wn0 = (const float*)d_in[5];
    const float* b0  = (const float*)d_in[6];
    const float* Ws1 = (const float*)d_in[7];
    const float* Wn1 = (const float*)d_in[8];
    const float* b1  = (const float*)d_in[9];
    const float* Ws2 = (const float*)d_in[10];
    const float* Wn2 = (const float*)d_in[11];
    const float* b2  = (const float*)d_in[12];
    const float* Wr  = (const float*)d_in[13];
    const float* br  = (const float*)d_in[14];
    float* out = (float*)d_out;

    // Carve workspace (~48 MB), 256B-aligned buffers.
    char* w = (char*)d_ws;
    auto carve = [&](size_t bytes) -> char* {
        char* p = w;
        w += (bytes + 255) & ~(size_t)255;
        return p;
    };
    float*    deg   = (float*)carve((size_t)N * 4);            // -> deg_inv in place
    float*    h_cur = (float*)carve((size_t)N * 64 * 4);       // f32 activations
    float*    agg   = (float*)carve((size_t)N * 64 * 4);       // scatter target
    _Float16* a1    = (_Float16*)carve((size_t)N * 128 * 2);   // [self64|neigh64] f16
    float*    h0    = (float*)carve((size_t)N * 16 * 4);       // padded layer-0 input
    float*    agg0  = (float*)carve((size_t)N * 16 * 4);
    _Float16* a0    = (_Float16*)carve((size_t)N * 32 * 2);    // [self16|neigh16] f16
    _Float16* wp0   = (_Float16*)carve((size_t)32 * 64 * 2);
    _Float16* wp1   = (_Float16*)carve((size_t)128 * 64 * 2);
    _Float16* wp2   = (_Float16*)carve((size_t)128 * 64 * 2);

    const int B = 256;
    const int ntiles = N / 16;                     // 3125 exactly
    const int gemm_blocks = (ntiles * 32 + B - 1) / B;

    // degree -> 1/max(deg,1)
    gnn_zero_f32<<<(N + B - 1) / B, B, 0, stream>>>(deg, N);
    gnn_edge_deg<<<(E + B - 1) / B, B, 0, stream>>>(dst, deg, E);
    gnn_invert_deg<<<(N + B - 1) / B, B, 0, stream>>>(deg, N);

    // layer-0 input + packed weights
    gnn_build_h0<<<(N * 16 + B - 1) / B, B, 0, stream>>>(features, h0, a0, N);
    gnn_pack_w<<<(32 * 64 + B - 1) / B, B, 0, stream>>>(Ws0, Wn0, wp0, 16, 10, 32);
    gnn_pack_w<<<(128 * 64 + B - 1) / B, B, 0, stream>>>(Ws1, Wn1, wp1, 64, 64, 128);
    gnn_pack_w<<<(128 * 64 + B - 1) / B, B, 0, stream>>>(Ws2, Wn2, wp2, 64, 64, 128);

    // ---- layer 0 (dims padded 10 -> 16) ----
    gnn_zero_f32<<<(N * 16 + B - 1) / B, B, 0, stream>>>(agg0, N * 16);
    gnn_edge_scatter<16><<<(E * 4 + B - 1) / B, B, 0, stream>>>(src, dst, e_feat, h0, agg0, E);
    gnn_pack_neigh<<<(N * 16 + B - 1) / B, B, 0, stream>>>(agg0, deg, a0, 16, 32, N);
    gnn_sage_gemm<1, true><<<gemm_blocks, B, 0, stream>>>(a0, 32, wp0, b0,
                                                          h_cur, a1, 128, ntiles);

    // ---- layer 1 ----
    gnn_zero_f32<<<(N * 64 + B - 1) / B, B, 0, stream>>>(agg, N * 64);
    gnn_edge_scatter<64><<<(E * 16 + B - 1) / B, B, 0, stream>>>(src, dst, e_feat, h_cur, agg, E);
    gnn_pack_neigh<<<(N * 64 + B - 1) / B, B, 0, stream>>>(agg, deg, a1, 64, 128, N);
    gnn_sage_gemm<4, true><<<gemm_blocks, B, 0, stream>>>(a1, 128, wp1, b1,
                                                          h_cur, a1, 128, ntiles);

    // ---- layer 2 (no sigmoid) ----
    gnn_zero_f32<<<(N * 64 + B - 1) / B, B, 0, stream>>>(agg, N * 64);
    gnn_edge_scatter<64><<<(E * 16 + B - 1) / B, B, 0, stream>>>(src, dst, e_feat, h_cur, agg, E);
    gnn_pack_neigh<<<(N * 64 + B - 1) / B, B, 0, stream>>>(agg, deg, a1, 64, 128, N);
    gnn_sage_gemm<4, false><<<gemm_blocks, B, 0, stream>>>(a1, 128, wp2, b2,
                                                           h_cur, (_Float16*)nullptr, 0, ntiles);

    // ---- readout ----
    gnn_readout<<<(N + B - 1) / B, B, 0, stream>>>(features, h_cur, Wr, br, out, N);
}
